// LAFDescriptor_47820165874295
// MI455X (gfx1250) — matrix operand order repaired and checked by reading, not compiled, not run on hardware
//
#include <hip/hip_runtime.h>
#include <hip/hip_bf16.h>

typedef _Float16 h16;
typedef _Float16 v16h __attribute__((ext_vector_type(16)));
typedef _Float16 v8h  __attribute__((ext_vector_type(8)));
typedef float    v8f  __attribute__((ext_vector_type(8)));
typedef unsigned int u32x4 __attribute__((ext_vector_type(4)));
typedef int          i32x8 __attribute__((ext_vector_type(8)));
typedef int          i32x4 __attribute__((ext_vector_type(4)));

#define BNSCALE 0.99999500003749969f  // 1/sqrt(1+1e-5)

__device__ __forceinline__ v16h cat8(v8h a, v8h b) {
    union { v16h v; v8h h[2]; } u; u.h[0] = a; u.h[1] = b; return u.v;
}

__device__ __forceinline__ v8f wmma16(v16h a, v16h b, v8f c) {
    return __builtin_amdgcn_wmma_f32_16x16x32_f16(false, a, false, b, (short)0, c, false, false);
}

// TDM: DMA a 1-row tile of `nelem` f16 from global `gsrc` into LDS byte
// offset `ldsoff` (relative to the wave's LDS base).  ISA 08_async_tensor §8.
// amdgpu-toolchain (clang-23) 6-arg builtin form.
__device__ __forceinline__ void tdm_load_row(const void* gsrc, unsigned ldsoff, int nelem) {
    unsigned long long ga = (unsigned long long)gsrc;
    u32x4 g0 = { 1u,                                      // count=1, user mode
                 ldsoff,                                  // lds_addr (bytes)
                 (unsigned)ga,                            // global_addr[31:0]
                 (unsigned)((ga >> 32) & 0x1FFFFFFu) | (2u << 30) };  // addr[56:32] | type=2
    i32x8 g1 = { (int)(1u << 16),                         // data_size=1 (2B), mask=0
                 (int)((nelem & 0xFFFF) << 16),           // tensor_dim0[15:0]
                 (int)(((unsigned)nelem >> 16) | (1u << 16)), // tensor_dim0[31:16] | tensor_dim1[15:0]=1
                 (int)((unsigned)nelem << 16),            // tensor_dim1[31:16]=0 | tile_dim0
                 1,                                       // tile_dim1=1, tile_dim2=0
                 nelem,                                   // tensor_dim0_stride[31:0]
                 (int)((nelem & 0xFFFF) << 16),           // str0[47:32]=0 | tensor_dim1_stride[15:0]
                 (int)((unsigned)nelem >> 16) };          // tensor_dim1_stride[47:16]
    i32x4 z4 = { 0, 0, 0, 0 };
    i32x8 z8 = { 0, 0, 0, 0, 0, 0, 0, 0 };
    __builtin_amdgcn_tensor_load_to_lds(g0, g1, z4, z4, z8, 0);
}

// ---------------------------------------------------------------------------
// Grayscale: (2,3,1024,1024) f32 -> (2,1024,1024) f32
// ---------------------------------------------------------------------------
__global__ __launch_bounds__(256) void k_gray(const float* __restrict__ img,
                                              float* __restrict__ gray) {
    int i = blockIdx.x * 256 + threadIdx.x;
    int b = i >> 20, p = i & 1048575;
    const float* base = img + b * 3 * 1048576;
    gray[i] = 0.299f * base[p] + 0.587f * base[1048576 + p] + 0.114f * base[2 * 1048576 + p];
}

// ---------------------------------------------------------------------------
// pyrdown: blur(reflect)+2x2avg == separable [1,5,10,10,5,1]/32, stride 2
// ---------------------------------------------------------------------------
__global__ __launch_bounds__(256) void k_pyrdown(const float* __restrict__ src,
                                                 float* __restrict__ dst, int H, int W) {
    int HO = H >> 1, WO = W >> 1;
    int i = blockIdx.x * 256 + threadIdx.x;
    int tot = 2 * HO * WO;
    if (i >= tot) return;
    int b = i / (HO * WO), r = i - b * HO * WO;
    int oy = r / WO, ox = r - oy * WO;
    const float* s = src + b * H * W;
    const float wt[6] = {1.f, 5.f, 10.f, 10.f, 5.f, 1.f};
    int ry[6], rx[6];
#pragma unroll
    for (int j = 0; j < 6; ++j) {
        int ty = 2 * oy - 2 + j; ty = ty < 0 ? -ty : (ty >= H ? 2 * H - 2 - ty : ty); ry[j] = ty;
        int tx = 2 * ox - 2 + j; tx = tx < 0 ? -tx : (tx >= W ? 2 * W - 2 - tx : tx); rx[j] = tx;
    }
    float acc = 0.f;
#pragma unroll
    for (int jy = 0; jy < 6; ++jy) {
        float rowa = 0.f;
        const float* row = s + ry[jy] * W;
#pragma unroll
        for (int jx = 0; jx < 6; ++jx) rowa += wt[jx] * row[rx[jx]];
        acc += wt[jy] * rowa;
    }
    dst[i] = acc * (1.f / 1024.f);
}

// ---------------------------------------------------------------------------
// Weight prep, emitted directly in LDS staging order:
//   conv w (O,C,3,3) f32 -> f16 [tap][C/8][O][8]   (c = grp*8 + cl)
//   w7 (128,128,8,8) f32 -> f16 [o][k'] , k' = flat of [c/8][s][8]  (s=y*8+x)
// ---------------------------------------------------------------------------
__device__ __forceinline__ void conv_tr(const float* __restrict__ s, h16* __restrict__ d,
                                        int O, int C, int i) {
    int cl = i & 7;
    int o = (i >> 3) % O;
    int r = (i >> 3) / O;          // t*(C/8) + grp   (O, C/8 are powers of 2)
    int grp = r % (C >> 3);
    int t = r / (C >> 3);
    int c = grp * 8 + cl;
    d[i] = (h16)s[(o * C + c) * 9 + t];
}

__global__ __launch_bounds__(256) void k_prep(
    const float* __restrict__ w2, const float* __restrict__ w3, const float* __restrict__ w4,
    const float* __restrict__ w5, const float* __restrict__ w6, const float* __restrict__ w7,
    h16* __restrict__ t2, h16* __restrict__ t3, h16* __restrict__ t4,
    h16* __restrict__ t5, h16* __restrict__ t6, h16* __restrict__ t7) {
    int i = blockIdx.x * 256 + threadIdx.x;
    if (i < 9216) { conv_tr(w2, t2, 32, 32, i); return; }     i -= 9216;
    if (i < 18432) { conv_tr(w3, t3, 64, 32, i); return; }    i -= 18432;
    if (i < 36864) { conv_tr(w4, t4, 64, 64, i); return; }    i -= 36864;
    if (i < 73728) { conv_tr(w5, t5, 128, 64, i); return; }   i -= 73728;
    if (i < 147456) { conv_tr(w6, t6, 128, 128, i); return; } i -= 147456;
    if (i < 1048576) {
        int o = i >> 13, k = i & 8191;
        int grp = k >> 9, sidx = (k >> 3) & 63, cl = k & 7;
        int c = grp * 8 + cl;
        t7[i] = (h16)w7[((o << 7) + c) * 64 + sidx];
    }
}

// ---------------------------------------------------------------------------
// Generic 3x3 conv, implicit GEMM on v_wmma_f32_16x16x32_f16.
//   M = Cout (one 16-tile per wave), N = out pixels, K = Cin x 9 taps.
//   Features LDS layout: [Cin/8][NPIX+1][8] f16; pixel NPIX is a zero slot
//   used branchlessly for padding. Weights DMA'd per tap by the TDM into
//   [Cin/8][Cout][8].  All fragment accesses are 16B with 16B lane stride.
// ---------------------------------------------------------------------------
template <int CIN, int COUT, int HIN, int S>
__device__ void conv3x3(const h16* __restrict__ wtaps, const h16* fin, h16* fout,
                        h16* wlds, unsigned wlds_off, int tid) {
    constexpr int HO = HIN / S;
    constexpr int NPO = HO * HO;
    constexpr int NPIN = HIN * HIN;
    constexpr int FSI = NPIN + 1;           // padded pixel stride (in)
    constexpr int FSO = NPO + 1;            // padded pixel stride (out)
    constexpr int NT = NPO / 16;
    constexpr int TILES = (COUT / 16) * NT;
    constexpr int PW = TILES / 8;           // tiles per wave
    constexpr int KC = CIN / 32;            // K chunks per tap
    constexpr int WSZ = COUT * CIN;
    const int wave = tid >> 5, lane = tid & 31;
    const int g = lane >> 4, l = lane & 15;
    const int tm = (wave * PW) / NT;        // all this wave's tiles share tm
    const int tn0 = (wave * PW) % NT;
    const int o = tm * 16 + l;              // A-fragment row for this lane

    int pyS[PW], pxS[PW], pbase[PW];
#pragma unroll
    for (int t = 0; t < PW; ++t) {
        const int p = (tn0 + t) * 16 + l;
        const int py = p / HO, px = p - py * HO;
        pyS[t] = py * S; pxS[t] = px * S;
        pbase[t] = py * S * HIN + px * S;
    }

    v8f acc[PW] = {};

    for (int tap = 0; tap < 9; ++tap) {
        const int dy = tap / 3 - 1, dx = tap - (tap / 3) * 3 - 1;
        const int dpix = dy * HIN + dx;
        __syncthreads();                    // prev tap's wlds reads done
        if (wave == 0) {                    // TDM: async tile DMA global->LDS
            tdm_load_row(wtaps + tap * WSZ, wlds_off, WSZ);
            __builtin_amdgcn_s_wait_tensorcnt(0);
        }
        __syncthreads();
        // A fragments: shared by every tile of this wave
        v16h A[KC];
#pragma unroll
        for (int kc = 0; kc < KC; ++kc) {
            v8h a0 = *(const v8h*)(wlds + (((kc * 4 + 2 * g) * COUT) + o) * 8);
            v8h a1 = *(const v8h*)(wlds + (((kc * 4 + 2 * g + 1) * COUT) + o) * 8);
            A[kc] = cat8(a0, a1);
        }
#pragma unroll
        for (int t = 0; t < PW; ++t) {
            const bool inb = (unsigned)(pyS[t] + dy) < (unsigned)HIN &&
                             (unsigned)(pxS[t] + dx) < (unsigned)HIN;
            const int pin = inb ? (pbase[t] + dpix) : NPIN;   // zero slot if OOB
#pragma unroll
            for (int kc = 0; kc < KC; ++kc) {
                const h16* bp = fin + (((kc * 4 + 2 * g) * FSI) + pin) * 8;
                v8h b0 = *(const v8h*)bp;
                v8h b1 = *(const v8h*)(bp + FSI * 8);
                acc[t] = wmma16(A[kc], cat8(b0, b1), acc[t]);
            }
        }
    }
    __syncthreads();
#pragma unroll
    for (int t = 0; t < PW; ++t) {
        const int p = (tn0 + t) * 16 + l;
        v8h r;
#pragma unroll
        for (int j = 0; j < 8; ++j) r[j] = (h16)fmaxf(acc[t][j] * BNSCALE, 0.f);
        *(v8h*)(fout + (((2 * tm + g) * FSO) + p) * 8) = r;   // 8 channels/lane, 16B
    }
    if (tid < COUT) fout[(((tid >> 3) * FSO) + NPO) * 8 + (tid & 7)] = (h16)0.f;
    __syncthreads();
}

// ---------------------------------------------------------------------------
// Mega kernel: per patch (2048 WG x 256 thr): extraction + std-norm + L1(VALU)
// + L2..L6 (WMMA, LDS resident) -> g6 f16 [patch][8192].
// Dynamic LDS: bufA 66048 | bufB 66048 | wlds 32768  (164864 B total).
// ---------------------------------------------------------------------------
__global__ __launch_bounds__(256) void k_hardnet(
    const float* __restrict__ lafs,
    const float* __restrict__ g0, const float* __restrict__ g1, const float* __restrict__ g2,
    const float* __restrict__ w1,
    const h16* __restrict__ wt2, const h16* __restrict__ wt3, const h16* __restrict__ wt4,
    const h16* __restrict__ wt5, const h16* __restrict__ wt6,
    h16* __restrict__ g6out) {
    extern __shared__ char smem[];
    h16* bufA = (h16*)smem;                          // [C/8][NPIX+1][8] views
    h16* bufB = (h16*)(smem + 66048);
    h16* wlds = (h16*)(smem + 132096);               // 32768 B (TDM target)
    float* pf32 = (float*)bufB;                      // 4096 B, dead before L2
    float* wsm  = (float*)(smem + 66048 + 4096);     // 1152 B (w1)
    float* red  = (float*)(smem + 66048 + 4096 + 1152);
    const unsigned wlds_off = (unsigned)__builtin_amdgcn_groupstaticsize() + 132096u;

    const int tid = threadIdx.x;
    const int patch = blockIdx.x;

    // ---- LAF & pyramid level ----
    const float a00 = lafs[patch * 6 + 0], a01 = lafs[patch * 6 + 1], cx = lafs[patch * 6 + 2];
    const float a10 = lafs[patch * 6 + 3], a11 = lafs[patch * 6 + 4], cy = lafs[patch * 6 + 5];
    const float scale = 2.f * sqrtf(fabsf(a00 * a11 - a01 * a10)) * (1.f / 32.f);
    int lvl = (int)fmaxf(log2f(scale), 0.f);
    lvl = lvl > 2 ? 2 : lvl;
    const int b = patch >> 10;
    const float* src; int dim; float inv2l;
    if (lvl == 0)      { src = g0 + b * 1048576; dim = 1024; inv2l = 1.f;   }
    else if (lvl == 1) { src = g1 + b * 262144;  dim = 512;  inv2l = 0.5f;  }
    else               { src = g2 + b * 65536;   dim = 256;  inv2l = 0.25f; }

    // ---- bilinear sample 32x32 (border clamp, align_corners=False) ----
    float vloc[4]; float s1 = 0.f, s2 = 0.f;
#pragma unroll
    for (int k = 0; k < 4; ++k) {
        int e = k * 256 + tid;
        int r = e >> 5, c = e & 31;
        float u = -1.f + c * (2.f / 31.f);
        float v = -1.f + r * (2.f / 31.f);
        float pxx = (a00 * u + a01 * v + cx) * inv2l;
        float pyy = (a10 * u + a11 * v + cy) * inv2l;
        float gx = fminf(fmaxf(pxx - 0.5f, 0.f), dim - 1.f);
        float gy = fminf(fmaxf(pyy - 0.5f, 0.f), dim - 1.f);
        int x0 = (int)gx, y0 = (int)gy;
        float wx = gx - x0, wy = gy - y0;
        int x1 = min(x0 + 1, dim - 1), y1 = min(y0 + 1, dim - 1);
        float v00 = src[y0 * dim + x0], v01 = src[y0 * dim + x1];
        float v10 = src[y1 * dim + x0], v11 = src[y1 * dim + x1];
        float val = v00 * (1 - wx) * (1 - wy) + v01 * wx * (1 - wy)
                  + v10 * (1 - wx) * wy + v11 * wx * wy;
        vloc[k] = val; s1 += val; s2 += val * val;
    }
    for (int off = 16; off > 0; off >>= 1) {
        s1 += __shfl_down(s1, off, 32);
        s2 += __shfl_down(s2, off, 32);
    }
    if ((tid & 31) == 0) { red[tid >> 5] = s1; red[8 + (tid >> 5)] = s2; }
    if (tid < 288) wsm[tid] = w1[tid];
    __syncthreads();
    if (tid == 0) {
        float t1 = 0.f, t2 = 0.f;
        for (int i = 0; i < 8; ++i) { t1 += red[i]; t2 += red[8 + i]; }
        float mu = t1 * (1.f / 1024.f);
        float var = (t2 - 1024.f * mu * mu) * (1.f / 1023.f);
        red[16] = mu; red[17] = sqrtf(fmaxf(var, 0.f)) + 1e-6f;
    }
    __syncthreads();
    const float mu = red[16], isd = 1.f / red[17];
#pragma unroll
    for (int k = 0; k < 4; ++k) { int e = k * 256 + tid; pf32[e] = (vloc[k] - mu) * isd; }
    __syncthreads();

    // ---- L1: 1->32 ch VALU conv -> bufA [4][1025][8] f16 ----
    for (int e = tid; e < 32768; e += 256) {
        int c = e & 31, pix = e >> 5;
        int y = pix >> 5, x = pix & 31;
        float s = 0.f;
#pragma unroll
        for (int t = 0; t < 9; ++t) {
            int iy = y + t / 3 - 1, ix = x + t % 3 - 1;
            float pv = ((unsigned)iy < 32u && (unsigned)ix < 32u) ? pf32[iy * 32 + ix] : 0.f;
            s += wsm[c * 9 + t] * pv;
        }
        bufA[((c >> 3) * 1025 + pix) * 8 + (c & 7)] = (h16)fmaxf(s * BNSCALE, 0.f);
    }
    if (tid < 32) bufA[((tid >> 3) * 1025 + 1024) * 8 + (tid & 7)] = (h16)0.f;  // zero slot
    __syncthreads();

    // ---- L2..L6 on WMMA ----
    conv3x3<32, 32, 32, 1>(wt2, bufA, bufB, wlds, wlds_off, tid);
    conv3x3<32, 64, 32, 2>(wt3, bufB, bufA, wlds, wlds_off, tid);
    conv3x3<64, 64, 16, 1>(wt4, bufA, bufB, wlds, wlds_off, tid);
    conv3x3<64, 128, 16, 2>(wt5, bufB, bufA, wlds, wlds_off, tid);
    conv3x3<128, 128, 8, 1>(wt6, bufA, bufB, wlds, wlds_off, tid);

    // ---- L6 activations -> g6 (strip the per-group zero-slot padding) ----
    for (int i = tid * 8; i < 8192; i += 2048) {
        int grp = i >> 9, rem = i & 511;
        *(v8h*)(g6out + (size_t)patch * 8192 + i) = *(const v8h*)(bufB + grp * 65 * 8 + rem);
    }
}

// ---------------------------------------------------------------------------
// L7 batched GEMM: M=128 ch, N=2048 patches, K=8192; fused BN + L2-normalize.
// g6 + w7t (34 MB f16) stay resident in the 192 MB L2.
// ---------------------------------------------------------------------------
__global__ __launch_bounds__(256) void k_l7(const h16* __restrict__ g6,
                                            const h16* __restrict__ w7t,
                                            float* __restrict__ out) {
    __shared__ h16 desc[128 * 128];
    __shared__ float invn[128];
    const int tid = threadIdx.x;
    const int wave = tid >> 5, lane = tid & 31;
    const int g = lane >> 4, l = lane & 15;
    const int pbase = blockIdx.x * 128;
    const int tm = wave;

    v8f acc[8] = {};
    const h16* arow = w7t + (tm * 16 + l) * 8192 + (g ? 8 : 0);
    for (int kc = 0; kc < 256; ++kc) {
        const int kb = kc * 32;
        if (kc + 1 < 256) {
            __builtin_prefetch(arow + kb + 32, 0, 1);
            __builtin_prefetch(g6 + (size_t)(pbase + l) * 8192 + kb + 32, 0, 1);
        }
        v8h a0 = *(const v8h*)(arow + kb);
        v8h a1 = *(const v8h*)(arow + kb + 16);
        v16h A = cat8(a0, a1);
#pragma unroll
        for (int tn = 0; tn < 8; ++tn) {
            const h16* brow = g6 + (size_t)(pbase + tn * 16 + l) * 8192 + kb + 8 * g;
            v8h b0 = *(const v8h*)brow;
            v8h b1 = *(const v8h*)(brow + 16);
            acc[tn] = wmma16(A, cat8(b0, b1), acc[tn]);
        }
    }
#pragma unroll
    for (int tn = 0; tn < 8; ++tn) {
        int p = tn * 16 + l;
        v8h r;
#pragma unroll
        for (int j = 0; j < 8; ++j) r[j] = (h16)(acc[tn][j] * BNSCALE);  // no ReLU on L7
        *(v8h*)(desc + p * 128 + tm * 16 + 8 * g) = r;
    }
    __syncthreads();
    if (tid < 128) {
        float s = 0.f;
        for (int o = 0; o < 128; ++o) { float v = (float)desc[tid * 128 + o]; s += v * v; }
        invn[tid] = 1.f / fmaxf(sqrtf(s), 1e-12f);
    }
    __syncthreads();
    for (int e = tid; e < 16384; e += 256) {
        int p = e >> 7, o = e & 127;
        out[(pbase + p) * 128 + o] = (float)desc[p * 128 + o] * invn[p];
    }
}

// ---------------------------------------------------------------------------
extern "C" void kernel_launch(void* const* d_in, const int* in_sizes, int n_in,
                              void* d_out, int out_size, void* d_ws, size_t ws_size,
                              hipStream_t stream) {
    const float* img  = (const float*)d_in[0];
    const float* lafs = (const float*)d_in[1];
    const float* w1 = (const float*)d_in[2];
    const float* w2 = (const float*)d_in[3];
    const float* w3 = (const float*)d_in[4];
    const float* w4 = (const float*)d_in[5];
    const float* w5 = (const float*)d_in[6];
    const float* w6 = (const float*)d_in[7];
    const float* w7 = (const float*)d_in[8];

    char* ws = (char*)d_ws;
    float* g0 = (float*)(ws + 0);                 // 8388608 B
    float* g1 = (float*)(ws + 8388608);           // 2097152 B
    float* g2 = (float*)(ws + 10485760);          // 524288 B
    h16* t2  = (h16*)(ws + 11010048);
    h16* t3  = (h16*)(ws + 11028480);
    h16* t4  = (h16*)(ws + 11065344);
    h16* t5  = (h16*)(ws + 11139072);
    h16* t6  = (h16*)(ws + 11286528);
    h16* t7  = (h16*)(ws + 11581440);
    h16* g6  = (h16*)(ws + 13678592);             // 33554432 B (end ~47.2 MB)

    k_gray<<<8192, 256, 0, stream>>>(img, g0);
    k_pyrdown<<<(2 * 512 * 512 + 255) / 256, 256, 0, stream>>>(g0, g1, 1024, 1024);
    k_pyrdown<<<(2 * 256 * 256 + 255) / 256, 256, 0, stream>>>(g1, g2, 512, 512);
    k_prep<<<5212, 256, 0, stream>>>(w2, w3, w4, w5, w6, w7, t2, t3, t4, t5, t6, t7);
    k_hardnet<<<2048, 256, 164864, stream>>>(lafs, g0, g1, g2, w1, t2, t3, t4, t5, t6, g6);
    k_l7<<<16, 256, 0, stream>>>(g6, t7, (float*)d_out);
}
